// NAT_Global_Transformer_81853486727757
// MI455X (gfx1250) — compile-verified, matrix operand order checked by reading, not run on hardware
//
#include <hip/hip_runtime.h>
#include <hip/hip_bf16.h>

// ---------------------------------------------------------------------------
// NAT transformer block for gfx1250 (MI455X), bf16 WMMA everywhere a GEMM is.
// ---------------------------------------------------------------------------

typedef __attribute__((ext_vector_type(16))) __bf16 v16bf;
typedef __attribute__((ext_vector_type(8)))  float  v8f;

#define BM 128
#define BN 64
#define BK 32
#define ASTR 40   // padded LDS row stride in bf16 elements (80B = 20 banks)

// --- WMMA fragment load from an LDS tile stored row-major [row][k], stride ASTR.
// ISA 16-bit A layout (wave32): lanes 0-15 => M=lane, lanes 16-31 => M=lane-16
// with K offset +8; VGPR j holds K pair (j>=4 ? 16:0) + 2*(j&3).
__device__ __forceinline__ v16bf frag_ld(const __bf16* buf, int rowbase, int lane) {
  const __bf16* p = buf + (size_t)(rowbase + (lane & 15)) * ASTR + ((lane >> 4) << 3);
  v16bf f;
#pragma unroll
  for (int j = 0; j < 8; ++j) {
    int k = ((j & 4) << 2) + ((j & 3) << 1);
    f[2 * j]     = p[k];
    f[2 * j + 1] = p[k + 1];
  }
  return f;
}

// ===========================================================================
// Generic GEMM:  C[M,N] = epilogue( A_bf16[M,K] * W_f32[K,N] )
// epilogue: v = (acc + bias[n]) * alpha ; optional exact GELU ; optional
// residual add ; store fp32 (optionally NCHW-transposed) and/or bf16.
// Block: 256 threads = 8 waves. Tile BM=128 x BN=64, wave w -> rows w*16..+16,
// all 64 cols (4 x v_wmma_f32_16x16x32_bf16 per K step).
// M % 128 == 0, N % 64 == 0, K % 32 == 0 for all calls here.
// ===========================================================================
__global__ __launch_bounds__(256) void gemm_bf16_wmma(
    const __bf16* __restrict__ A, const float* __restrict__ Wt,
    const float* __restrict__ bias, const float* __restrict__ resid,
    float alpha, int act_gelu,
    float* __restrict__ Cf, __bf16* __restrict__ Cb,
    int M, int N, int K, int nchw, int HW) {
  __shared__ __bf16 As[BM * ASTR];
  __shared__ __bf16 Bs[BN * ASTR];   // stored transposed: [n][k]

  const int nblk = N / BN;
  const int m0 = (blockIdx.x / nblk) * BM;
  const int n0 = (blockIdx.x % nblk) * BN;
  const int tid  = threadIdx.x;
  const int lane = tid & 31;
  const int wave = tid >> 5;

  v8f acc[4];
#pragma unroll
  for (int i = 0; i < 4; ++i)
#pragma unroll
    for (int j = 0; j < 8; ++j) acc[i][j] = 0.f;

  for (int k0 = 0; k0 < K; k0 += BK) {
    // ---- A tile: 128x32 bf16, vectorized 16B loads/stores
    {
      int r = tid >> 1;
      int c = (tid & 1) << 4;                    // 0 or 16
      const uint4* s4 = reinterpret_cast<const uint4*>(A + (size_t)(m0 + r) * K + k0 + c);
      uint4* d4 = reinterpret_cast<uint4*>(&As[r * ASTR + c]);
      d4[0] = s4[0];
      d4[1] = s4[1];
      if (k0 + BK < K)  // pull next A tile toward L2 (global_prefetch_b8)
        __builtin_prefetch(A + (size_t)(m0 + r) * K + k0 + BK + c, 0, 3);
    }
    // ---- B tile: W is K x N fp32 row-major; convert and store [n][k]
    {
      int n  = tid >> 2;
      int kk = (tid & 3) << 3;
#pragma unroll
      for (int j = 0; j < 8; ++j) {
        float w = Wt[(size_t)(k0 + kk + j) * N + n0 + n];
        Bs[n * ASTR + kk + j] = (__bf16)w;
      }
    }
    __syncthreads();

    v16bf a = frag_ld(As, wave * 16, lane);
#pragma unroll
    for (int nt = 0; nt < 4; ++nt) {
      v16bf b = frag_ld(Bs, nt * 16, lane);
      acc[nt] = __builtin_amdgcn_wmma_f32_16x16x32_bf16(
          false, a, false, b, (short)0, acc[nt], false, false);
    }
    __syncthreads();
  }

  // ---- Epilogue. D layout: n = lane&15, m = r + (lane>=16 ? 8 : 0)
  const int nloc = lane & 15;
  const int moff = (lane >> 4) << 3;
#pragma unroll
  for (int nt = 0; nt < 4; ++nt) {
    int n = n0 + nt * 16 + nloc;
    float bi = bias ? bias[n] : 0.f;
#pragma unroll
    for (int r = 0; r < 8; ++r) {
      int m = m0 + wave * 16 + moff + r;
      float v = (acc[nt][r] + bi) * alpha;
      if (act_gelu) v = 0.5f * v * (1.f + erff(v * 0.70710678118f));
      if (resid) v += resid[(size_t)m * N + n];
      if (Cf) {
        if (nchw) {
          int b   = m / HW;
          int rem = m - b * HW;
          Cf[((size_t)b * N + n) * HW + rem] = v;
        } else {
          Cf[(size_t)m * N + n] = v;
        }
      }
      if (Cb) Cb[(size_t)m * N + n] = (__bf16)v;
    }
  }
}

// ===========================================================================
// 3x3 conv (pad 1) as implicit GEMM: M = B*H*W, K = Cin*9, N = Cout.
// OIHW weights: k = ci*9 + ky*3 + kx is contiguous per output channel.
// Output NHWC fp32 (+bias).
// ===========================================================================
__global__ __launch_bounds__(256) void conv3x3_wmma(
    const float* __restrict__ x, const float* __restrict__ w,
    const float* __restrict__ bias, float* __restrict__ y,
    int Cin, int Cout, int H, int Wd) {
  __shared__ __bf16 As[BM * ASTR];
  __shared__ __bf16 Bs[BN * ASTR];

  const int K = Cin * 9;
  const int nblk = Cout / BN;
  const int m0 = (blockIdx.x / nblk) * BM;
  const int n0 = (blockIdx.x % nblk) * BN;
  const int tid  = threadIdx.x;
  const int lane = tid & 31;
  const int wave = tid >> 5;
  const int HWp  = H * Wd;

  v8f acc[4];
#pragma unroll
  for (int i = 0; i < 4; ++i)
#pragma unroll
    for (int j = 0; j < 8; ++j) acc[i][j] = 0.f;

  for (int k0 = 0; k0 < K; k0 += BK) {
    // A tile: im2col gather with zero padding
    for (int i = tid; i < BM * BK; i += 256) {
      int r = i >> 5, c = i & 31;
      int m = m0 + r, k = k0 + c;
      int b   = m / HWp;
      int rem = m - b * HWp;
      int yy  = rem / Wd;
      int xx  = rem - yy * Wd;
      int ci  = k / 9;
      int p   = k - ci * 9;
      int ky  = p / 3, kx = p - ky * 3;
      int iy = yy + ky - 1, ix = xx + kx - 1;
      float v = 0.f;
      if (iy >= 0 && iy < H && ix >= 0 && ix < Wd)
        v = x[(((size_t)b * Cin + ci) * H + iy) * Wd + ix];
      As[r * ASTR + c] = (__bf16)v;
    }
    // B tile transposed [n][k]; OIHW is contiguous in k
    for (int i = tid; i < BN * BK; i += 256) {
      int n = i >> 5, c = i & 31;
      Bs[n * ASTR + c] = (__bf16)w[(size_t)(n0 + n) * K + k0 + c];
    }
    __syncthreads();

    v16bf a = frag_ld(As, wave * 16, lane);
#pragma unroll
    for (int nt = 0; nt < 4; ++nt) {
      v16bf b = frag_ld(Bs, nt * 16, lane);
      acc[nt] = __builtin_amdgcn_wmma_f32_16x16x32_bf16(
          false, a, false, b, (short)0, acc[nt], false, false);
    }
    __syncthreads();
  }

  const int nloc = lane & 15;
  const int moff = (lane >> 4) << 3;
#pragma unroll
  for (int nt = 0; nt < 4; ++nt) {
    int n = n0 + nt * 16 + nloc;
    float bi = bias[n];
#pragma unroll
    for (int r = 0; r < 8; ++r) {
      int m = m0 + wave * 16 + moff + r;
      y[(size_t)m * Cout + n] = acc[nt][r] + bi;
    }
  }
}

// ===========================================================================
// Dual LayerNorm over C=256: one wave per row, lane owns 8 strided channels.
// Writes up to two affine-transformed bf16 outputs from one stats pass.
// ===========================================================================
__global__ __launch_bounds__(256) void ln_dual(
    const float* __restrict__ x, int M,
    const float* __restrict__ w1, const float* __restrict__ b1, __bf16* __restrict__ o1,
    const float* __restrict__ w2, const float* __restrict__ b2, __bf16* __restrict__ o2) {
  const int C = 256;
  int row  = blockIdx.x * 8 + (threadIdx.x >> 5);
  int lane = threadIdx.x & 31;
  if (row >= M) return;
  const float* xr = x + (size_t)row * C;
  float vals[8];
  float s = 0.f;
#pragma unroll
  for (int j = 0; j < 8; ++j) { vals[j] = xr[lane + 32 * j]; s += vals[j]; }
#pragma unroll
  for (int off = 16; off > 0; off >>= 1) s += __shfl_xor(s, off, 32);
  float mu = s * (1.f / C);
  float var = 0.f;
#pragma unroll
  for (int j = 0; j < 8; ++j) { float d = vals[j] - mu; var += d * d; }
#pragma unroll
  for (int off = 16; off > 0; off >>= 1) var += __shfl_xor(var, off, 32);
  float rstd = rsqrtf(var * (1.f / C) + 1e-5f);
#pragma unroll
  for (int j = 0; j < 8; ++j) {
    int c = lane + 32 * j;
    float nz = (vals[j] - mu) * rstd;
    o1[(size_t)row * C + c] = (__bf16)(nz * w1[c] + b1[c]);
    if (o2) o2[(size_t)row * C + c] = (__bf16)(nz * w2[c] + b2[c]);
  }
}

// ===========================================================================
// NAT 3x3-window cross attention. One thread per (pixel, head); hd = 32.
// q pre-scaled by hd^-0.5 in its projection GEMM. kv packs k (ch 0..255) and
// v (ch 256..511). Output bf16 (feeds the wp projection GEMM as its A).
// ===========================================================================
__global__ __launch_bounds__(256) void nat_attn(
    const __bf16* __restrict__ q, const __bf16* __restrict__ kv,
    const float* __restrict__ rpb, __bf16* __restrict__ out,
    int Bn, int H, int Wd) {
  int idx = blockIdx.x * 256 + threadIdx.x;
  int total = Bn * H * Wd * 8;
  if (idx >= total) return;
  int head = idx & 7;
  int m    = idx >> 3;
  int HWp  = H * Wd;
  int b    = m / HWp;
  int rem  = m - b * HWp;
  int y    = rem / Wd;
  int x    = rem - y * Wd;

  int si = y - 1; si = si < 0 ? 0 : si; if (si > H - 3) si = H - 3;
  int sj = x - 1; sj = sj < 0 ? 0 : sj; if (sj > Wd - 3) sj = Wd - 3;
  int pi = y - si, pj = x - sj;

  float qv[32];
  const __bf16* qp = q + (size_t)m * 256 + head * 32;
#pragma unroll
  for (int d = 0; d < 32; ++d) qv[d] = (float)qp[d];

  float logit[9];
  float mx = -1e30f;
#pragma unroll
  for (int a = 0; a < 3; ++a) {
#pragma unroll
    for (int bb = 0; bb < 3; ++bb) {
      int mm = (b * H + si + a) * Wd + sj + bb;
      const __bf16* kp = kv + (size_t)mm * 512 + head * 32;
      float dot = 0.f;
#pragma unroll
      for (int d = 0; d < 32; ++d) dot += qv[d] * (float)kp[d];
      int ii = a - pi + 2, jj = bb - pj + 2;
      float l = dot + rpb[(head * 5 + ii) * 5 + jj];
      logit[a * 3 + bb] = l;
      mx = fmaxf(mx, l);
    }
  }

  float accv[32];
#pragma unroll
  for (int d = 0; d < 32; ++d) accv[d] = 0.f;
  float denom = 0.f;
#pragma unroll
  for (int a = 0; a < 3; ++a) {
#pragma unroll
    for (int bb = 0; bb < 3; ++bb) {
      float wgt = __expf(logit[a * 3 + bb] - mx);
      denom += wgt;
      int mm = (b * H + si + a) * Wd + sj + bb;
      const __bf16* vp = kv + (size_t)mm * 512 + 256 + head * 32;
#pragma unroll
      for (int d = 0; d < 32; ++d) accv[d] += wgt * (float)vp[d];
    }
  }
  float inv = 1.f / denom;
  __bf16* op = out + (size_t)m * 256 + head * 32;
#pragma unroll
  for (int d = 0; d < 32; ++d) op[d] = (__bf16)(accv[d] * inv);
}

// ===========================================================================
// Host orchestration
// ===========================================================================
extern "C" void kernel_launch(void* const* d_in, const int* in_sizes, int n_in,
                              void* d_out, int out_size, void* d_ws, size_t ws_size,
                              hipStream_t stream) {
  (void)in_sizes; (void)n_in; (void)out_size; (void)ws_size;
  const int Bn = 2, IMG = 112, CIN = 64, C = 256;
  const size_t Mpix = (size_t)Bn * IMG * IMG;       // 25088 (multiple of 128)
  const int HW = IMG * IMG;

  const float* xq      = (const float*)d_in[0];
  const float* xkv     = (const float*)d_in[1];
  const float* conv1_w = (const float*)d_in[2];
  const float* conv1_b = (const float*)d_in[3];
  const float* conv3_w = (const float*)d_in[4];
  const float* conv3_b = (const float*)d_in[5];
  const float* ln1_w   = (const float*)d_in[6];
  const float* ln1_b   = (const float*)d_in[7];
  const float* ln10_w  = (const float*)d_in[8];
  const float* ln10_b  = (const float*)d_in[9];
  const float* ln3_w   = (const float*)d_in[10];
  const float* ln3_b   = (const float*)d_in[11];
  const float* wq      = (const float*)d_in[12];
  const float* bq      = (const float*)d_in[13];
  const float* wkv     = (const float*)d_in[14];
  const float* bkv     = (const float*)d_in[15];
  const float* rpb     = (const float*)d_in[16];
  const float* wp      = (const float*)d_in[17];
  const float* bp      = (const float*)d_in[18];
  const float* fc1_w   = (const float*)d_in[19];
  const float* fc1_b   = (const float*)d_in[20];
  const float* fc2_w   = (const float*)d_in[21];
  const float* fc2_b   = (const float*)d_in[22];

  // ---- workspace arena (with reuse). szF = Mpix*256*4 ; szH = Mpix*256*2
  char* ws = (char*)d_ws;
  const size_t szF  = Mpix * 256 * sizeof(float);   // 25.7 MB
  const size_t szH  = Mpix * 256 * sizeof(short);   // 12.8 MB
  const size_t szH2 = Mpix * 512 * sizeof(short);   // 25.7 MB

  float*  x1     = (float*)(ws);                         // [0, szF)
  float*  x3     = (float*)(ws + szF);                   // [szF, 2szF)
  __bf16* qn     = (__bf16*)(ws + 2 * szF);              // [2szF, +szH)
  __bf16* kvn    = (__bf16*)(ws + 2 * szF + szH);        // [.., +szH)
  __bf16* qb     = (__bf16*)(ws);                        // reuse x1 slot
  __bf16* kvb    = (__bf16*)(ws + 2 * szF + 2 * szH);    // [.., +szH2)
  __bf16* attout = (__bf16*)(ws + 2 * szF);              // reuse qn slot
  float*  att    = (float*)(ws + 2 * szF + 2 * szH);     // reuse kvb slot
  __bf16* hbuf   = (__bf16*)(ws);                        // reuse qb slot
  __bf16* tbuf   = (__bf16*)(ws + szF);                  // reuse x3 slot
  float*  outp   = (float*)d_out;

  const int mb = (int)(Mpix / BM);                       // 196
  const float qscale = 0.17677669529663689f;             // 32^-0.5

  // 1/2: convs -> x1, x3 (NHWC fp32)
  conv3x3_wmma<<<dim3(mb * (C / BN)), dim3(256), 0, stream>>>(
      xq, conv1_w, conv1_b, x1, CIN, C, IMG, IMG);
  conv3x3_wmma<<<dim3(mb * (C / BN)), dim3(256), 0, stream>>>(
      xkv, conv3_w, conv3_b, x3, CIN, C, IMG, IMG);

  // 3: dual LN of x1 -> qn (ln1), kvn (ln10)
  ln_dual<<<dim3((int)(Mpix / 8)), dim3(256), 0, stream>>>(
      x1, (int)Mpix, ln1_w, ln1_b, qn, ln10_w, ln10_b, kvn);

  // 4: q = (qn*wq + bq) * scale  -> bf16
  gemm_bf16_wmma<<<dim3(mb * (C / BN)), dim3(256), 0, stream>>>(
      qn, wq, bq, nullptr, qscale, 0, nullptr, qb,
      (int)Mpix, C, C, 0, HW);
  // 5: kv = kvn*wkv + bkv -> bf16 (k || v)
  gemm_bf16_wmma<<<dim3(mb * (2 * C / BN)), dim3(256), 0, stream>>>(
      kvn, wkv, bkv, nullptr, 1.f, 0, nullptr, kvb,
      (int)Mpix, 2 * C, C, 0, HW);

  // 6: neighborhood attention -> attout bf16
  nat_attn<<<dim3((int)(Mpix * 8 / 256)), dim3(256), 0, stream>>>(
      qb, kvb, rpb, attout, Bn, IMG, IMG);

  // 7: att = attout*wp + bp + x3 -> fp32
  gemm_bf16_wmma<<<dim3(mb * (C / BN)), dim3(256), 0, stream>>>(
      attout, wp, bp, x3, 1.f, 0, att, nullptr,
      (int)Mpix, C, C, 0, HW);

  // 8: h = LN(att, ln3) -> bf16
  ln_dual<<<dim3((int)(Mpix / 8)), dim3(256), 0, stream>>>(
      att, (int)Mpix, ln3_w, ln3_b, hbuf, nullptr, nullptr, nullptr);

  // 9: t = gelu(h*fc1 + b1) -> bf16
  gemm_bf16_wmma<<<dim3(mb * (2 * C / BN)), dim3(256), 0, stream>>>(
      hbuf, fc1_w, fc1_b, nullptr, 1.f, 1, nullptr, tbuf,
      (int)Mpix, 2 * C, C, 0, HW);

  // 10: out = (t*fc2 + b2) + att, stored NCHW fp32
  gemm_bf16_wmma<<<dim3(mb * (C / BN)), dim3(256), 0, stream>>>(
      tbuf, fc2_w, fc2_b, att, 1.f, 0, outp, nullptr,
      (int)Mpix, C, 2 * C, 1, HW);
}
